// LLaMABaseline_17265768530801
// MI455X (gfx1250) — compile-verified, matrix operand order checked by reading, not run on hardware
//
#include <hip/hip_runtime.h>
#include <math.h>

// ---- model dims (match reference) ----
#define LNUM   8
#define DMODEL 1024
#define NH     16
#define NKV    4
#define GRP    4          // NH / NKV
#define DHEAD  64
#define FF     2816
#define VOCAB  32000
#define TSEQ   1024
#define BBATCH 2
#define NTOK   (BBATCH*TSEQ)   // 2048
#define EPSV   1e-6f
#define SCALEV 0.125f          // DHEAD^-0.5

typedef __attribute__((ext_vector_type(16))) __bf16 v16bf;
typedef __attribute__((ext_vector_type(8)))  float  v8f;

union FragAB { v16bf v; uint4 u[2]; };

#define LDSROW 40                 // 32 K-elems + 8 pad (80B rows, 16B aligned)
#define TILEEL (128 * LDSROW)     // elements per tile buffer

// ======================================================================
// Generic NT GEMM: C[M,N](f32) = A[M,K](bf16,row-major) * W[N,K](bf16)^T
// Double-buffered gfx1250 async global->LDS pipeline (ASYNCcnt) feeding
// v_wmma_f32_16x16x32_bf16. Block tile 128x128x32, 8 waves, 64x32/wave.
// ======================================================================
__global__ __launch_bounds__(256)
void gemm_bf16_nt(const __bf16* __restrict__ A, const __bf16* __restrict__ W,
                  const float* __restrict__ Res, float* __restrict__ C,
                  int M, int N, int K,
                  long aStride,                 // A batch stride (elements)
                  int wDiv, long wStride,       // W offset = (bz/wDiv)*wStride
                  int cDiv, long cS1, long cS2, // C offset = (bz/cDiv)*cS1 + (bz%cDiv)*cS2
                  int ldc, int causalSkip)
{
  const int bz    = blockIdx.z;
  const int mBase = blockIdx.y * 128;
  const int nBase = blockIdx.x * 128;
  if (causalSkip && (nBase > mBase + 127)) return;   // fully above diagonal

  const __bf16* Ab = A + (long)bz * aStride;
  const __bf16* Wb = W + (long)(bz / wDiv) * wStride;
  const long cBase = (long)(bz / cDiv) * cS1 + (long)(bz % cDiv) * cS2;

  __shared__ __attribute__((aligned(16))) __bf16 sA[2][TILEEL];
  __shared__ __attribute__((aligned(16))) __bf16 sB[2][TILEEL];

  const int tid  = threadIdx.x;
  const int wave = tid >> 5;
  const int lane = tid & 31;
  const int wm = wave >> 2;        // 0..1  (M direction)
  const int wn = wave & 3;         // 0..3  (N direction)
  const int laneRow  = lane & 15;
  const int laneHalf = lane >> 4;

  // per-thread tile-load geometry: 2 chunks of 16B for A, same for B
  const int chunk0 = tid;                    // 0..255
  const int chunk1 = tid + 256;              // 256..511
  const int r0 = chunk0 >> 2, c0 = (chunk0 & 3) << 3;
  const int r1 = chunk1 >> 2, c1 = (chunk1 & 3) << 3;
  const bool aOk0 = (mBase + r0) < M, aOk1 = (mBase + r1) < M;
  const bool bOk0 = (nBase + r0) < N, bOk1 = (nBase + r1) < N;
  const __bf16* aP0 = Ab + (long)(mBase + r0) * K + c0;
  const __bf16* aP1 = Ab + (long)(mBase + r1) * K + c1;
  const __bf16* bP0 = Wb + (long)(nBase + r0) * K + c0;
  const __bf16* bP1 = Wb + (long)(nBase + r1) * K + c1;
  const unsigned laBase = (unsigned)(size_t)&sA[0][0];
  const unsigned lbBase = (unsigned)(size_t)&sB[0][0];

  // issue async global->LDS loads for K-tile at k0 into buffer `par`
  auto issueTile = [&](int k0, int par) {
    unsigned la0 = laBase + (unsigned)(par * TILEEL + r0 * LDSROW + c0) * 2u;
    unsigned la1 = laBase + (unsigned)(par * TILEEL + r1 * LDSROW + c1) * 2u;
    unsigned lb0 = lbBase + (unsigned)(par * TILEEL + r0 * LDSROW + c0) * 2u;
    unsigned lb1 = lbBase + (unsigned)(par * TILEEL + r1 * LDSROW + c1) * 2u;
    if (aOk0)
      asm volatile("global_load_async_to_lds_b128 %0, %1, off"
                   :: "v"(la0), "v"((unsigned long long)(size_t)(aP0 + k0)) : "memory");
    if (aOk1)
      asm volatile("global_load_async_to_lds_b128 %0, %1, off"
                   :: "v"(la1), "v"((unsigned long long)(size_t)(aP1 + k0)) : "memory");
    if (bOk0)
      asm volatile("global_load_async_to_lds_b128 %0, %1, off"
                   :: "v"(lb0), "v"((unsigned long long)(size_t)(bP0 + k0)) : "memory");
    if (bOk1)
      asm volatile("global_load_async_to_lds_b128 %0, %1, off"
                   :: "v"(lb1), "v"((unsigned long long)(size_t)(bP1 + k0)) : "memory");
  };

  v8f acc[4][2];
#pragma unroll
  for (int i = 0; i < 4; ++i)
#pragma unroll
    for (int j = 0; j < 2; ++j)
#pragma unroll
      for (int r = 0; r < 8; ++r) acc[i][j][r] = 0.0f;

  const int nk = K >> 5;
  issueTile(0, 0);

  for (int kt = 0; kt < nk; ++kt) {
    const int par = kt & 1;
    // our async writes of tile kt complete; barrier also retires all waves'
    // LDS reads of the other buffer from iteration kt-1.
    asm volatile("s_wait_asynccnt 0x0" ::: "memory");
    __syncthreads();
    if (kt + 1 < nk) issueTile((kt + 1) << 5, par ^ 1);   // overlap with compute

    // ---- LDS -> fragments (16-bit A 16x32 layout; B identical, W is NxK) ----
    FragAB fa[4], fb[2];
#pragma unroll
    for (int i = 0; i < 4; ++i) {
      const __bf16* p = &sA[par][(wm * 64 + i * 16 + laneRow) * LDSROW + laneHalf * 8];
      fa[i].u[0] = *(const uint4*)p;
      fa[i].u[1] = *(const uint4*)(p + 16);
    }
#pragma unroll
    for (int j = 0; j < 2; ++j) {
      const __bf16* p = &sB[par][(wn * 32 + j * 16 + laneRow) * LDSROW + laneHalf * 8];
      fb[j].u[0] = *(const uint4*)p;
      fb[j].u[1] = *(const uint4*)(p + 16);
    }

#pragma unroll
    for (int i = 0; i < 4; ++i)
#pragma unroll
      for (int j = 0; j < 2; ++j)
        acc[i][j] = __builtin_amdgcn_wmma_f32_16x16x32_bf16(
            false, fa[i].v, false, fb[j].v, (short)0, acc[i][j], false, false);
  }

  // ---- store (C/D layout: lane=N col, vgpr r = M row within 8-row half) ----
#pragma unroll
  for (int j = 0; j < 2; ++j) {
    int col = nBase + wn * 32 + j * 16 + laneRow;
    if (col >= N) continue;
#pragma unroll
    for (int i = 0; i < 4; ++i) {
#pragma unroll
      for (int r = 0; r < 8; ++r) {
        int row = mBase + wm * 64 + i * 16 + laneHalf * 8 + r;
        if (row < M) {
          long off = cBase + (long)row * ldc + col;
          float v = acc[i][j][r];
          if (Res) v += Res[off];
          C[off] = v;
        }
      }
    }
  }
}

// ======================================================================
// Elementwise / small kernels
// ======================================================================
__global__ void cvt_f32_bf16_k(const float* __restrict__ in, __bf16* __restrict__ out, long n)
{
  long i = (long)blockIdx.x * blockDim.x + threadIdx.x;
  if (i < n) out[i] = (__bf16)in[i];
}

__global__ void embed_k(const int* __restrict__ ids, const float* __restrict__ emb,
                        float* __restrict__ x)
{
  long i = (long)blockIdx.x * blockDim.x + threadIdx.x;
  if (i >= (long)NTOK * DMODEL) return;
  int d  = (int)(i % DMODEL);
  int bt = (int)(i / DMODEL);
  x[i] = emb[(long)ids[bt] * DMODEL + d];
}

__global__ __launch_bounds__(256)
void rmsnorm_k(const float* __restrict__ x, const float* __restrict__ w,
               __bf16* __restrict__ out)
{
  __shared__ float red[256];
  int row = blockIdx.x, tid = threadIdx.x;
  const float* xr = x + (long)row * DMODEL;
  float acc = 0.f;
  for (int i = tid; i < DMODEL; i += 256) { float v = xr[i]; acc += v * v; }
  red[tid] = acc; __syncthreads();
  for (int s = 128; s > 0; s >>= 1) {
    if (tid < s) red[tid] += red[tid + s];
    __syncthreads();
  }
  float rinv = rsqrtf(red[0] / (float)DMODEL + EPSV);
  __bf16* orow = out + (long)row * DMODEL;
  for (int i = tid; i < DMODEL; i += 256) orow[i] = (__bf16)(xr[i] * rinv * w[i]);
}

// RoPE: in f32 [B,T,nH,DH] -> out bf16 [B,nH,T,DH]   (one thread per pair)
__global__ void rope_k(const float* __restrict__ in, __bf16* __restrict__ out, int nH)
{
  long gid = (long)blockIdx.x * blockDim.x + threadIdx.x;
  long total = (long)BBATCH * TSEQ * nH * (DHEAD / 2);
  if (gid >= total) return;
  int p = (int)(gid & 31); long tmp = gid >> 5;
  int h = (int)(tmp % nH); tmp /= nH;
  int t = (int)(tmp % TSEQ);
  int b = (int)(tmp / TSEQ);
  float inv = __powf(10000.0f, -(float)p / 32.0f);   // theta^{-2p/DH}
  float s, c; __sincosf((float)t * inv, &s, &c);
  long io = (((long)b * TSEQ + t) * nH + h) * DHEAD + 2 * p;
  float x0 = in[io], x1 = in[io + 1];
  long oo = (((long)b * nH + h) * TSEQ + t) * DHEAD + 2 * p;
  out[oo]     = (__bf16)(x0 * c - x1 * s);
  out[oo + 1] = (__bf16)(x0 * s + x1 * c);
}

// V: f32 [B,T,NKV,DH] -> bf16 [B,NKV,DH,T]  (transposed so P.V is an NT GEMM)
__global__ void vtrans_k(const float* __restrict__ in, __bf16* __restrict__ out)
{
  long gid = (long)blockIdx.x * blockDim.x + threadIdx.x;
  if (gid >= (long)BBATCH * TSEQ * NKV * DHEAD) return;
  int d = (int)(gid % DHEAD); long tmp = gid / DHEAD;
  int h = (int)(tmp % NKV); tmp /= NKV;
  int t = (int)(tmp % TSEQ);
  int b = (int)(tmp / TSEQ);
  out[(((long)b * NKV + h) * DHEAD + d) * TSEQ + t] = (__bf16)in[gid];
}

// causal softmax over row of raw scores (scale fused); zero-fill masked tail
__global__ __launch_bounds__(256)
void softmax_k(const float* __restrict__ S, __bf16* __restrict__ P)
{
  __shared__ float red[256];
  int row = blockIdx.x, tid = threadIdx.x;
  int t = row & (TSEQ - 1);
  int len = t + 1;
  const float* sr = S + (long)row * TSEQ;
  float m = -3.4e38f;
  for (int s = tid; s < len; s += 256) m = fmaxf(m, sr[s] * SCALEV);
  red[tid] = m; __syncthreads();
  for (int s = 128; s > 0; s >>= 1) { if (tid < s) red[tid] = fmaxf(red[tid], red[tid + s]); __syncthreads(); }
  m = red[0]; __syncthreads();
  float sum = 0.f;
  for (int s = tid; s < len; s += 256) sum += __expf(sr[s] * SCALEV - m);
  red[tid] = sum; __syncthreads();
  for (int s = 128; s > 0; s >>= 1) { if (tid < s) red[tid] += red[tid + s]; __syncthreads(); }
  float rs = 1.0f / red[0];
  __bf16* pr = P + (long)row * TSEQ;
  for (int s = tid; s < TSEQ; s += 256)
    pr[s] = (s < len) ? (__bf16)(__expf(sr[s] * SCALEV - m) * rs) : (__bf16)0.0f;
}

__global__ void swiglu_k(const float* __restrict__ h1, const float* __restrict__ h3,
                         __bf16* __restrict__ out, long n)
{
  long i = (long)blockIdx.x * blockDim.x + threadIdx.x;
  if (i >= n) return;
  float a = h1[i];
  float g = a / (1.0f + __expf(-a));
  out[i] = (__bf16)(g * h3[i]);
}

// ======================================================================
// Host orchestration
// ======================================================================
extern "C" void kernel_launch(void* const* d_in, const int* in_sizes, int n_in,
                              void* d_out, int out_size, void* d_ws, size_t ws_size,
                              hipStream_t stream)
{
  const int*   ids = (const int*)  d_in[0];
  const float* emb = (const float*)d_in[1];
  const float* qw  = (const float*)d_in[2];
  const float* kw  = (const float*)d_in[3];
  const float* vw  = (const float*)d_in[4];
  const float* ow  = (const float*)d_in[5];
  const float* w1  = (const float*)d_in[6];
  const float* w2  = (const float*)d_in[7];
  const float* w3  = (const float*)d_in[8];
  const float* n1  = (const float*)d_in[9];
  const float* n2  = (const float*)d_in[10];
  const float* nf  = (const float*)d_in[11];
  float* logits = (float*)d_out;

  char* base = (char*)d_ws;
  size_t off = 0;
  auto alloc = [&](size_t bytes) -> void* {
    void* p = base + off;
    off += (bytes + 255) & ~(size_t)255;
    return p;
  };

  const long nQW = (long)LNUM * NH  * DHEAD * DMODEL;
  const long nKW = (long)LNUM * NKV * DHEAD * DMODEL;
  const long nOW = (long)LNUM * DMODEL * DMODEL;
  const long nFW = (long)LNUM * FF * DMODEL;
  const long nEM = (long)VOCAB * DMODEL;

  __bf16* bqw = (__bf16*)alloc(nQW * 2);
  __bf16* bkw = (__bf16*)alloc(nKW * 2);
  __bf16* bvw = (__bf16*)alloc(nKW * 2);
  __bf16* bow = (__bf16*)alloc(nOW * 2);
  __bf16* bw1 = (__bf16*)alloc(nFW * 2);
  __bf16* bw2 = (__bf16*)alloc(nFW * 2);
  __bf16* bw3 = (__bf16*)alloc(nFW * 2);
  __bf16* bem = (__bf16*)alloc(nEM * 2);

  float*  x    = (float*) alloc((long)NTOK * DMODEL * 4);
  __bf16* xn   = (__bf16*)alloc((long)NTOK * DMODEL * 2);
  float*  qf   = (float*) alloc((long)NTOK * NH  * DHEAD * 4);
  float*  kf   = (float*) alloc((long)NTOK * NKV * DHEAD * 4);
  float*  vf   = (float*) alloc((long)NTOK * NKV * DHEAD * 4);
  __bf16* qbf  = (__bf16*)alloc((long)NTOK * NH  * DHEAD * 2);
  __bf16* kbf  = (__bf16*)alloc((long)NTOK * NKV * DHEAD * 2);
  __bf16* vtbf = (__bf16*)alloc((long)NTOK * NKV * DHEAD * 2);
  float*  sc   = (float*) alloc((long)BBATCH * NH * TSEQ * TSEQ * 4);
  __bf16* pbf  = (__bf16*)alloc((long)BBATCH * NH * TSEQ * TSEQ * 2);
  float*  aof  = (float*) alloc((long)NTOK * DMODEL * 4);
  __bf16* aobf = (__bf16*)alloc((long)NTOK * DMODEL * 2);
  float*  h1   = (float*) alloc((long)NTOK * FF * 4);
  float*  h3   = (float*) alloc((long)NTOK * FF * 4);
  __bf16* hgbf = (__bf16*)alloc((long)NTOK * FF * 2);
  __bf16* xfbf = (__bf16*)alloc((long)NTOK * DMODEL * 2);

  auto cvt = [&](const float* in, __bf16* out, long n) {
    cvt_f32_bf16_k<<<(int)((n + 255) / 256), 256, 0, stream>>>(in, out, n);
  };
  cvt(qw, bqw, nQW); cvt(kw, bkw, nKW); cvt(vw, bvw, nKW); cvt(ow, bow, nOW);
  cvt(w1, bw1, nFW); cvt(w2, bw2, nFW); cvt(w3, bw3, nFW); cvt(emb, bem, nEM);

  auto gemm = [&](const __bf16* A, const __bf16* W, const float* Res, float* C,
                  int M, int N, int K, int batch,
                  long aStride, int wDiv, long wStride,
                  int cDiv, long cS1, long cS2, int ldc, int causal) {
    dim3 g((N + 127) / 128, (M + 127) / 128, batch);
    gemm_bf16_nt<<<g, 256, 0, stream>>>(A, W, Res, C, M, N, K,
                                        aStride, wDiv, wStride,
                                        cDiv, cS1, cS2, ldc, causal);
  };

  embed_k<<<(NTOK * DMODEL + 255) / 256, 256, 0, stream>>>(ids, emb, x);

  for (int l = 0; l < LNUM; ++l) {
    const __bf16* qwl = bqw + (long)l * NH  * DHEAD * DMODEL;
    const __bf16* kwl = bkw + (long)l * NKV * DHEAD * DMODEL;
    const __bf16* vwl = bvw + (long)l * NKV * DHEAD * DMODEL;
    const __bf16* owl = bow + (long)l * DMODEL * DMODEL;
    const __bf16* w1l = bw1 + (long)l * FF * DMODEL;
    const __bf16* w2l = bw2 + (long)l * FF * DMODEL;
    const __bf16* w3l = bw3 + (long)l * FF * DMODEL;

    // --- attention ---
    rmsnorm_k<<<NTOK, 256, 0, stream>>>(x, n1 + (long)l * DMODEL, xn);
    gemm(xn, qwl, nullptr, qf, NTOK, NH * DHEAD,  DMODEL, 1, 0, 1, 0, 1, 0, 0, NH * DHEAD, 0);
    gemm(xn, kwl, nullptr, kf, NTOK, NKV * DHEAD, DMODEL, 1, 0, 1, 0, 1, 0, 0, NKV * DHEAD, 0);
    gemm(xn, vwl, nullptr, vf, NTOK, NKV * DHEAD, DMODEL, 1, 0, 1, 0, 1, 0, 0, NKV * DHEAD, 0);
    rope_k<<<(BBATCH * TSEQ * NH * 32 + 255) / 256, 256, 0, stream>>>(qf, qbf, NH);
    rope_k<<<(BBATCH * TSEQ * NKV * 32 + 255) / 256, 256, 0, stream>>>(kf, kbf, NKV);
    vtrans_k<<<(BBATCH * TSEQ * NKV * DHEAD + 255) / 256, 256, 0, stream>>>(vf, vtbf);

    // scores[b,h] = q[b,h] . k[b,h/G]^T  (causal block-skip)
    gemm(qbf, kbf, nullptr, sc, TSEQ, TSEQ, DHEAD, BBATCH * NH,
         (long)TSEQ * DHEAD, GRP, (long)TSEQ * DHEAD,
         1, (long)TSEQ * TSEQ, 0, TSEQ, 1);
    softmax_k<<<BBATCH * NH * TSEQ, 256, 0, stream>>>(sc, pbf);
    // out[b,t,h*64+d] = P[b,h] . V[b,h/G]   (Vt stored [b,kv,DH,T])
    gemm(pbf, vtbf, nullptr, aof, TSEQ, DHEAD, TSEQ, BBATCH * NH,
         (long)TSEQ * TSEQ, GRP, (long)DHEAD * TSEQ,
         NH, (long)TSEQ * DMODEL, DHEAD, DMODEL, 0);
    cvt(aof, aobf, (long)NTOK * DMODEL);
    // x += attn_out . ow^T
    gemm(aobf, owl, x, x, NTOK, DMODEL, DMODEL, 1, 0, 1, 0, 1, 0, 0, DMODEL, 0);

    // --- MLP ---
    rmsnorm_k<<<NTOK, 256, 0, stream>>>(x, n2 + (long)l * DMODEL, xn);
    gemm(xn, w1l, nullptr, h1, NTOK, FF, DMODEL, 1, 0, 1, 0, 1, 0, 0, FF, 0);
    gemm(xn, w3l, nullptr, h3, NTOK, FF, DMODEL, 1, 0, 1, 0, 1, 0, 0, FF, 0);
    swiglu_k<<<(int)(((long)NTOK * FF + 255) / 256), 256, 0, stream>>>(h1, h3, hgbf, (long)NTOK * FF);
    gemm(hgbf, w2l, x, x, NTOK, DMODEL, FF, 1, 0, 1, 0, 1, 0, 0, DMODEL, 0);
  }

  rmsnorm_k<<<NTOK, 256, 0, stream>>>(x, nf, xfbf);
  gemm(xfbf, bem, nullptr, logits, NTOK, VOCAB, DMODEL, 1, 0, 1, 0, 1, 0, 0, VOCAB, 0);
}
